// GCNModel_2010044694697
// MI455X (gfx1250) — matrix operand order repaired and checked by reading, not compiled, not run on hardware
//
#include <hip/hip_runtime.h>

typedef __attribute__((ext_vector_type(16))) __bf16 v16bf;
typedef __attribute__((ext_vector_type(8)))  float  v8f;

#define BN_EPS 1e-5f

__device__ __forceinline__ unsigned short f2bf(float f) {
  unsigned u = __float_as_uint(f);
  // round-to-nearest-even bf16
  return (unsigned short)((u + 0x7FFFu + ((u >> 16) & 1u)) >> 16);
}

// ---------------------------------------------------------------- converts
__global__ void cvt_bf16_kernel(const float* __restrict__ src,
                                unsigned short* __restrict__ dst, long n) {
  long i = (long)blockIdx.x * blockDim.x + threadIdx.x;
  if (i < n) dst[i] = f2bf(src[i]);
}

__global__ void zero_f32_kernel(float* __restrict__ p, long n) {
  long i = (long)blockIdx.x * blockDim.x + threadIdx.x;
  if (i < n) p[i] = 0.f;
}

// W[K x H] row-major f32  ->  Wt[H x K] "column-major" bf16 (row n holds column n of W)
__global__ void wt_cvt_kernel(const float* __restrict__ W,
                              unsigned short* __restrict__ Wt, int K, int H) {
  long i = (long)blockIdx.x * blockDim.x + threadIdx.x;
  if (i >= (long)K * H) return;
  int n = (int)(i % H);
  int k = (int)(i / H);
  Wt[(long)n * K + k] = f2bf(W[(long)k * H + n]);
}

// ---------------------------------------------------------------- WMMA GEMM
// C[N x H] = A[N x K](bf16) @ W[K x H] + bias, optional ReLU, f32 out.
// One wave computes a 16(M) x 64(N) tile: 1 A-fragment reused across 4 B-fragments.
// All fragment loads for a k-step are issued before the 4 back-to-back WMMAs so
// later loads overlap earlier matrix ops.
// Requires: N%16==0, K%32==0, H%64==0.
union FragAB { v16bf v; uint4 q[2]; };

__global__ __launch_bounds__(256) void gemm_wmma_kernel(
    const unsigned short* __restrict__ A, const unsigned short* __restrict__ Wt,
    const float* __restrict__ bias, float* __restrict__ C,
    int N, int K, int H, int relu) {
  int lane = threadIdx.x & 31;
  int wave = blockIdx.x * (blockDim.x >> 5) + (threadIdx.x >> 5);
  int nt64s = H >> 6;
  if (wave >= (N >> 4) * nt64s) return;
  int mt = wave / nt64s;
  int nt = wave % nt64s;
  int half = lane >> 4;     // lane group selects K sub-range per ISA layout
  int l15  = lane & 15;

  // A fragment base: row m = mt*16 + l15; dwords 0..3 -> K = half*8 + [0..7],
  // dwords 4..7 -> K = 16 + half*8 + [0..7]  (16-bit A 16x32 layout)
  const unsigned short* arow = A + (long)(mt * 16 + l15) * K + half * 8;
  // B fragment base per tile t: col n, dwords 0..3 -> K = half*8 + [0..7], dwords 4..7 -> K = 16 + half*8 + [0..7]
  const unsigned short* brow = Wt + (long)(nt * 64 + l15) * K + half * 8;

  v8f zero = {0.f, 0.f, 0.f, 0.f, 0.f, 0.f, 0.f, 0.f};
  v8f acc[4];
#pragma unroll
  for (int t = 0; t < 4; ++t) acc[t] = zero;

  for (int k0 = 0; k0 < K; k0 += 32) {
    FragAB a, b[4];
    a.q[0] = *(const uint4*)(arow + k0);        // K = k0 + half*8 + 0..7
    a.q[1] = *(const uint4*)(arow + k0 + 16);   // K = k0 + 16 + half*8 + 0..7
#pragma unroll
    for (int t = 0; t < 4; ++t) {
      const unsigned short* bp = brow + (long)(t * 16) * K + k0;
      b[t].q[0] = *(const uint4*)bp;
      b[t].q[1] = *(const uint4*)(bp + 16);
    }
#pragma unroll
    for (int t = 0; t < 4; ++t)
      acc[t] = __builtin_amdgcn_wmma_f32_16x16x32_bf16(
          false, a.v, false, b[t].v, (short)0, acc[t], false, false);
  }

#pragma unroll
  for (int t = 0; t < 4; ++t) {
    int n = (nt * 4 + t) * 16 + l15;
    float bv = bias[n];
#pragma unroll
    for (int i = 0; i < 8; ++i) {
      int m = mt * 16 + i + half * 8;        // C/D layout: VGPR i -> row i (+8 for hi half)
      float r = acc[t][i] + bv;
      if (relu) r = fmaxf(r, 0.f);
      C[(long)m * H + n] = r;
    }
  }
}

// ---------------------------------------------------------------- edge scatter
// One wave per EPW edges: agg[rows[e], :] += vals[e] * t[cols[e], :]
// float4 coalesced gathers from L2-resident t, hardware f32 atomics into agg.
// Next edge's source row is prefetched (global_prefetch) while atomics drain.
#define EPW 4

__global__ __launch_bounds__(256) void scatter_edges_kernel(
    const int* __restrict__ rows, const int* __restrict__ cols,
    const float* __restrict__ vals, const float* __restrict__ t,
    float* __restrict__ agg, long E, int H) {
  long w = ((long)blockIdx.x * blockDim.x + threadIdx.x) >> 5;
  int lane = threadIdx.x & 31;
  long e0 = w * EPW;
  if (e0 >= E) return;

  // warm up: prefetch first edge's source row (one 128B line per lane)
  if (lane * 32 < H)
    __builtin_prefetch(t + (long)cols[e0] * H + lane * 32, 0, 1);

#pragma unroll
  for (int i = 0; i < EPW; ++i) {
    long e = e0 + i;
    if (e >= E) break;
    if (i + 1 < EPW && e + 1 < E && lane * 32 < H)
      __builtin_prefetch(t + (long)cols[e + 1] * H + lane * 32, 0, 1);
    int r = rows[e];
    int c = cols[e];
    float v = vals[e];
    const float4* src = (const float4*)(t + (long)c * H);
    float* dst = agg + (long)r * H;
    for (int j = lane; j * 4 < H; j += 32) {
      float4 s = src[j];
      int base = j * 4;
      unsafeAtomicAdd(dst + base + 0, v * s.x);
      unsafeAtomicAdd(dst + base + 1, v * s.y);
      unsafeAtomicAdd(dst + base + 2, v * s.z);
      unsafeAtomicAdd(dst + base + 3, v * s.w);
    }
  }
}

// ---------------------------------------------------------------- batchnorm
// stats[0..H-1] = column sums, stats[H..2H-1] = column sum of squares
__global__ void col_stats_kernel(const float* __restrict__ agg,
                                 float* __restrict__ stats, long N, int H) {
  int col = threadIdx.x;                 // blockDim.x == H (128 or 256)
  float s = 0.f, ss = 0.f;
  for (long r = blockIdx.y; r < N; r += gridDim.y) {
    float a = agg[r * H + col];
    s += a;
    ss += a * a;
  }
  unsafeAtomicAdd(&stats[col], s);
  unsafeAtomicAdd(&stats[H + col], ss);
}

__global__ void bn_relu_cvt_kernel(const float* __restrict__ agg,
                                   const float* __restrict__ stats,
                                   const float* __restrict__ g,
                                   const float* __restrict__ beta,
                                   unsigned short* __restrict__ out,
                                   long total, int H, float invN) {
  long i = (long)blockIdx.x * blockDim.x + threadIdx.x;
  if (i >= total) return;
  int col = (int)(i % H);
  float mu  = stats[col] * invN;
  float var = stats[H + col] * invN - mu * mu;
  float x = (agg[i] - mu) * rsqrtf(var + BN_EPS) * g[col] + beta[col];
  out[i] = f2bf(fmaxf(x, 0.f));
}

// ---------------------------------------------------------------- MLP head
// Each thread copies 8 bf16 (one b128) of the concatenated [u|v] row.
__global__ void gather_pairs_kernel(const unsigned short* __restrict__ x2,
                                    const int* __restrict__ ui,
                                    const int* __restrict__ vi,
                                    unsigned short* __restrict__ comb,
                                    int Bn, int h2, int n_users) {
  long i = (long)blockIdx.x * blockDim.x + threadIdx.x;
  int cw = (2 * h2) >> 3;                // 16B chunks per combined row
  long total = (long)Bn * cw;
  if (i >= total) return;
  int b = (int)(i / cw);
  int j = (int)(i % cw) << 3;            // element offset within combined row
  const unsigned short* src =
      (j < h2) ? (x2 + (long)ui[b] * h2 + j)
               : (x2 + (long)(n_users + vi[b]) * h2 + (j - h2));
  ((uint4*)comb)[i] = *(const uint4*)src;
}

__global__ void final_dot_kernel(const float* __restrict__ h,
                                 const float* __restrict__ P2,
                                 const float* __restrict__ pb2,
                                 float* __restrict__ out, int Bn, int h2) {
  int w = (int)(((long)blockIdx.x * blockDim.x + threadIdx.x) >> 5);
  int lane = threadIdx.x & 31;
  if (w >= Bn) return;
  float s = 0.f;
  for (int j = lane; j < h2; j += 32) s += h[(long)w * h2 + j] * P2[j];
#pragma unroll
  for (int off = 16; off > 0; off >>= 1) s += __shfl_xor(s, off, 32);
  if (lane == 0) out[w] = s + pb2[0];
}

// ---------------------------------------------------------------- launcher
#define CEILDIV(a, b) (((a) + (b) - 1) / (b))

extern "C" void kernel_launch(void* const* d_in, const int* in_sizes, int n_in,
                              void* d_out, int out_size, void* d_ws, size_t ws_size,
                              hipStream_t stream) {
  const int*   user_idx = (const int*)d_in[0];
  const int*   item_idx = (const int*)d_in[1];
  const int*   erows    = (const int*)d_in[2];
  const int*   ecols    = (const int*)d_in[3];
  const float* evals    = (const float*)d_in[4];
  const float* uemb     = (const float*)d_in[5];
  const float* iemb     = (const float*)d_in[6];
  const float* W1  = (const float*)d_in[7];  const float* b1  = (const float*)d_in[8];
  const float* g1  = (const float*)d_in[9];  const float* be1 = (const float*)d_in[10];
  const float* W2  = (const float*)d_in[11]; const float* b2  = (const float*)d_in[12];
  const float* g2  = (const float*)d_in[13]; const float* be2 = (const float*)d_in[14];
  const float* P1  = (const float*)d_in[15]; const float* pb1 = (const float*)d_in[16];
  const float* P2  = (const float*)d_in[17]; const float* pb2 = (const float*)d_in[18];

  const int  Bn = in_sizes[0];
  const long E  = in_sizes[2];
  const int  h1 = in_sizes[8];                 // 256
  const int  h2 = in_sizes[12];                // 128
  const int  d  = in_sizes[7] / h1;            // 128
  const int  n_users = in_sizes[5] / d;
  const int  n_items = in_sizes[6] / d;
  const long N = (long)n_users + n_items;      // 150000, %16==0

  // ---- workspace layout
  size_t off = 0;
  auto wsAlloc = [&](size_t bytes) -> void* {
    void* p = (char*)d_ws + off;
    off = (off + bytes + 255) & ~(size_t)255;
    return p;
  };
  unsigned short* bufA  = (unsigned short*)wsAlloc((size_t)N * h1 * sizeof(unsigned short));
  float*          bufT  = (float*)wsAlloc((size_t)N * h1 * sizeof(float));
  float*          bufAgg= (float*)wsAlloc((size_t)N * h1 * sizeof(float));
  unsigned short* wtb   = (unsigned short*)wsAlloc((size_t)2 * h1 * h2 * sizeof(unsigned short));
  float*          stats = (float*)wsAlloc((size_t)2 * h1 * sizeof(float));
  unsigned short* comb  = (unsigned short*)wsAlloc((size_t)Bn * 2 * h2 * sizeof(unsigned short));
  float*          hbuf  = (float*)wsAlloc((size_t)Bn * h2 * sizeof(float));

  auto gemm = [&](const unsigned short* A_, const unsigned short* Wt_,
                  const float* bias_, float* C_, long n_, int k_, int h_, int relu_) {
    long waves = (n_ >> 4) * (h_ >> 6);
    unsigned blocks = (unsigned)CEILDIV(waves * 32, 256);
    gemm_wmma_kernel<<<blocks, 256, 0, stream>>>(A_, Wt_, bias_, C_, (int)n_, k_, h_, relu_);
  };
  auto zero = [&](float* p, long n) {
    zero_f32_kernel<<<(unsigned)CEILDIV(n, 256), 256, 0, stream>>>(p, n);
  };
  auto scatter = [&](float* t_, float* agg_, int h_) {
    long waves = CEILDIV(E, (long)EPW);
    scatter_edges_kernel<<<(unsigned)CEILDIV(waves * 32, 256), 256, 0, stream>>>(
        erows, ecols, evals, t_, agg_, E, h_);
  };

  // ---- embeddings -> bf16 node features
  cvt_bf16_kernel<<<(unsigned)CEILDIV((long)n_users * d, 256), 256, 0, stream>>>(
      uemb, bufA, (long)n_users * d);
  cvt_bf16_kernel<<<(unsigned)CEILDIV((long)n_items * d, 256), 256, 0, stream>>>(
      iemb, bufA + (long)n_users * d, (long)n_items * d);

  // ================= GCN layer 1 =================
  wt_cvt_kernel<<<(unsigned)CEILDIV((long)d * h1, 256), 256, 0, stream>>>(W1, wtb, d, h1);
  gemm(bufA, wtb, b1, bufT, N, d, h1, 0);                       // t1 = x@W1+b1

  zero(bufAgg, N * h1);
  scatter(bufT, bufAgg, h1);

  zero(stats, 2 * h1);
  { dim3 grid(1, 1024);
    col_stats_kernel<<<grid, h1, 0, stream>>>(bufAgg, stats, N, h1); }
  bn_relu_cvt_kernel<<<(unsigned)CEILDIV(N * h1, 256), 256, 0, stream>>>(
      bufAgg, stats, g1, be1, bufA, N * h1, h1, 1.0f / (float)N); // x1 (bf16)

  // ================= GCN layer 2 =================
  wt_cvt_kernel<<<(unsigned)CEILDIV((long)h1 * h2, 256), 256, 0, stream>>>(W2, wtb, h1, h2);
  gemm(bufA, wtb, b2, bufT, N, h1, h2, 0);                      // t2 = x1@W2+b2

  zero(bufAgg, N * h2);
  scatter(bufT, bufAgg, h2);

  zero(stats, 2 * h2);
  { dim3 grid(1, 1024);
    col_stats_kernel<<<grid, h2, 0, stream>>>(bufAgg, stats, N, h2); }
  bn_relu_cvt_kernel<<<(unsigned)CEILDIV(N * h2, 256), 256, 0, stream>>>(
      bufAgg, stats, g2, be2, bufA, N * h2, h2, 1.0f / (float)N); // x2 (bf16)

  // ================= MLP head =================
  gather_pairs_kernel<<<(unsigned)CEILDIV((long)Bn * ((2 * h2) >> 3), 256), 256, 0, stream>>>(
      bufA, user_idx, item_idx, comb, Bn, h2, n_users);

  wt_cvt_kernel<<<(unsigned)CEILDIV((long)2 * h2 * h2, 256), 256, 0, stream>>>(P1, wtb, 2 * h2, h2);
  gemm(comb, wtb, pb1, hbuf, Bn, 2 * h2, h2, 1);                // h = relu(comb@P1+pb1)

  final_dot_kernel<<<(unsigned)CEILDIV((long)Bn * 32, 256), 256, 0, stream>>>(
      hbuf, P2, pb2, (float*)d_out, Bn, h2);
}
